// ModifiedMambaMixer_16355235463221
// MI455X (gfx1250) — matrix operand order repaired and checked by reading, not compiled, not run on hardware
//
#include <hip/hip_runtime.h>
#include <stdint.h>
#include <stddef.h>

// ---------------- problem constants ----------------
static constexpr int H_DIM = 2048;
static constexpr int DI    = 4096;
static constexpr int NST   = 16;     // SSM state dim
static constexpr int KC    = 4;      // conv taps
static constexpr int RR    = 128;    // dt rank
static constexpr int BB    = 2;
static constexpr int LL    = 1024;
static constexpr int PDIM  = RR + 2 * NST;  // 160
static constexpr int E2    = 2 * DI;        // 8192
static constexpr int LDK   = 40;     // padded LDS K-stride (halves): 80B row stride

// ---------------- types ----------------
typedef __bf16 bf16_t;
typedef bf16_t bf16x8  __attribute__((ext_vector_type(8)));
typedef bf16_t bf16x16 __attribute__((ext_vector_type(16)));
typedef float  f32x8   __attribute__((ext_vector_type(8)));
typedef int    v4i     __attribute__((vector_size(16)));

#ifndef __has_builtin
#define __has_builtin(x) 0
#endif
#if __has_builtin(__builtin_amdgcn_global_load_async_to_lds_b128)
#define HAVE_ASYNC_LDS 1
#else
#define HAVE_ASYNC_LDS 0
#endif

__device__ __forceinline__ uint16_t f32_to_bf16_bits(float f) {
  union { float f; uint32_t u; } v; v.f = f;
  uint32_t r = v.u + 0x7FFFu + ((v.u >> 16) & 1u);   // round-to-nearest-even
  return (uint16_t)(r >> 16);
}

__device__ __forceinline__ float silu_f(float x) {
  return x / (1.0f + __expf(-x));
}

// 16-byte global -> LDS copy (async DMA when available)
__device__ __forceinline__ void copy16(const uint16_t* g, uint16_t* l) {
#if HAVE_ASYNC_LDS
  // addrspacecast first (keeps element type), then reinterpret to v4i*
  __attribute__((address_space(1))) const uint16_t* g1 =
      (__attribute__((address_space(1))) const uint16_t*)g;
  __attribute__((address_space(3))) uint16_t* l3 =
      (__attribute__((address_space(3))) uint16_t*)l;
  __builtin_amdgcn_global_load_async_to_lds_b128(
      (__attribute__((address_space(1))) v4i*)g1,
      (__attribute__((address_space(3))) v4i*)l3, 0, 0);
#else
  *reinterpret_cast<uint4*>(l) = *reinterpret_cast<const uint4*>(g);
#endif
}

__device__ __forceinline__ void wait_async0() {
#if HAVE_ASYNC_LDS
#if __has_builtin(__builtin_amdgcn_s_wait_asynccnt)
  __builtin_amdgcn_s_wait_asynccnt(0);
#else
  asm volatile("s_wait_asynccnt 0x0" ::: "memory");
#endif
#endif
}

__device__ __forceinline__ void wait_async4() {
#if HAVE_ASYNC_LDS
#if __has_builtin(__builtin_amdgcn_s_wait_asynccnt)
  __builtin_amdgcn_s_wait_asynccnt(4);
#else
  asm volatile("s_wait_asynccnt 0x4" ::: "memory");
#endif
#endif
}

// ---------------- f32 -> bf16 convert ----------------
__global__ void cvt_f32_bf16(const float* __restrict__ src,
                             uint16_t* __restrict__ dst, int n) {
  int i = blockIdx.x * blockDim.x + threadIdx.x;
  if (i < n) dst[i] = f32_to_bf16_bits(src[i]);
}

// ---------------- WMMA GEMM with LDS double-buffering ----------------
// D[b][n][m] = sum_k A[m][k] * B[b][n][k]      (all K contiguous)
// A: M x K bf16 row-major (weights)
// B: N x K bf16 per batch (activations, k contiguous)
// D: N x M f32 per batch (m contiguous)
// EPI==1: D = softplus(acc + bias[m])
// Block tile 128(M) x 128(N); 8 waves, each 64x32; K-step 32, 2 LDS buffers.
template <int EPI>
__global__ __launch_bounds__(256)
void gemm_wmma_bf16(const uint16_t* __restrict__ Aq,
                    const uint16_t* __restrict__ Bq,
                    float* __restrict__ Dq,
                    const float* __restrict__ bias,
                    int M, int N, int K,
                    long strideB, long strideD) {
  __shared__ __align__(32) uint16_t lA[2][128 * LDK];   // 10 KB each
  __shared__ __align__(32) uint16_t lB[2][128 * LDK];   // 40 KB total

  const int b = blockIdx.z;
  const uint16_t* A  = Aq;
  const uint16_t* Bp = Bq + (size_t)b * strideB;
  float* Dp = Dq + (size_t)b * strideD;

  const int tid  = threadIdx.x;
  const int lane = tid & 31;
  const int wave = tid >> 5;
  const int wm = wave & 1;          // 2 waves in M
  const int wn = wave >> 1;         // 4 waves in N
  const int mW = wm * 64;           // wave tile inside the block tile
  const int nW = wn * 32;
  const int lr = lane & 15;
  const int lh = lane >> 4;

  // per-thread copy slice: 128 rows x 2 x 8 K-halves, 16B per thread
  const int cr = tid >> 1;          // 0..127
  const int cc = (tid & 1) * 16;    // 0 or 16 halves
  int arow = blockIdx.x * 128 + cr; if (arow >= M) arow = M - 1;
  const int brow = blockIdx.y * 128 + cr;          // N divisible by 128
  const uint16_t* aG = A  + (size_t)arow * K + cc;
  const uint16_t* bG = Bp + (size_t)brow * K + cc;
  uint16_t* aL[2] = { &lA[0][cr * LDK + cc], &lA[1][cr * LDK + cc] };
  uint16_t* bL[2] = { &lB[0][cr * LDK + cc], &lB[1][cr * LDK + cc] };

  f32x8 acc[4][2];
#pragma unroll
  for (int i = 0; i < 4; ++i)
#pragma unroll
    for (int j = 0; j < 2; ++j)
#pragma unroll
      for (int r = 0; r < 8; ++r) acc[i][j][r] = 0.0f;

  // prologue: fill buffer 0
  copy16(aG, aL[0]);
  copy16(bG, bL[0]);

  int cur = 0;
  for (int kk = 0; kk < K; kk += 32) {
    const bool hasNext = (kk + 32 < K);
    if (hasNext) {
      copy16(aG + kk + 32, aL[cur ^ 1]);
      copy16(bG + kk + 32, bL[cur ^ 1]);
      __builtin_prefetch(aG + kk + 96, 0, 1);      // pull K+64 toward L2
      wait_async4();                                // <=4 async ops left (a set is 2 ops)
    } else {
      wait_async0();
    }
    __syncthreads();                                // buf[cur] visible to all

    // B fragments: lane = col (wn*32 + j*16 + lr), 16 contiguous K at lh*16
    bf16x16 bfrag[2];
#pragma unroll
    for (int j = 0; j < 2; ++j) {
      const uint16_t* p = &lB[cur][(nW + j * 16 + lr) * LDK + lh * 16];
      bf16x8 b0 = *reinterpret_cast<const bf16x8*>(p);
      bf16x8 b1 = *reinterpret_cast<const bf16x8*>(p + 8);
#pragma unroll
      for (int t = 0; t < 8; ++t) { bfrag[j][t] = b0[t]; bfrag[j][t + 8] = b1[t]; }
    }
    // A fragments: lane = row (mW + i*16 + lr), 16 contiguous K at lh*16
    bf16x16 afrag[4];
#pragma unroll
    for (int i = 0; i < 4; ++i) {
      const uint16_t* p = &lA[cur][(mW + i * 16 + lr) * LDK + lh * 16];
      bf16x8 lo = *reinterpret_cast<const bf16x8*>(p);
      bf16x8 hi = *reinterpret_cast<const bf16x8*>(p + 8);
#pragma unroll
      for (int t = 0; t < 8; ++t) { afrag[i][t] = lo[t]; afrag[i][t + 8] = hi[t]; }
    }
#pragma unroll
    for (int i = 0; i < 4; ++i)
#pragma unroll
      for (int j = 0; j < 2; ++j)
        acc[i][j] = __builtin_amdgcn_wmma_f32_16x16x32_bf16(
            false, afrag[i], false, bfrag[j], (short)0, acc[i][j], false, false);

    __syncthreads();                // all reads of buf[cur] done before overwrite
    cur ^= 1;
  }

  // store: lane writes 8 contiguous M values at col n
  const int mB = blockIdx.x * 128;
  const int nB = blockIdx.y * 128;
#pragma unroll
  for (int i = 0; i < 4; ++i) {
#pragma unroll
    for (int j = 0; j < 2; ++j) {
      int m = mB + mW + i * 16 + lh * 8;
      if (m < M) {
        int n = nB + nW + j * 16 + lr;
        float v[8];
#pragma unroll
        for (int r = 0; r < 8; ++r) {
          float x = acc[i][j][r];
          if (EPI == 1) {
            x += bias[m + r];
            x = (x > 20.0f) ? x : log1pf(__expf(x));   // softplus
          }
          v[r] = x;
        }
        float4* o = reinterpret_cast<float4*>(Dp + (size_t)n * M + m);
        o[0] = make_float4(v[0], v[1], v[2], v[3]);
        o[1] = make_float4(v[4], v[5], v[6], v[7]);
      }
    }
  }
}

// ---------------- causal depthwise conv + SiLU ----------------
__global__ void conv_silu_kernel(const float* __restrict__ proj,
                                 const float* __restrict__ Wconv,
                                 const float* __restrict__ bconv,
                                 float* __restrict__ hsf,
                                 uint16_t* __restrict__ hsb) {
  const int d = blockIdx.x * blockDim.x + threadIdx.x;
  const int l = blockIdx.y;
  const int b = blockIdx.z;
  const float* pb = proj + (size_t)b * LL * E2;
  float acc = bconv[d];
#pragma unroll
  for (int k = 0; k < KC; ++k) {
    int li = l + k - (KC - 1);
    if (li >= 0) acc += Wconv[d * KC + k] * pb[(size_t)li * E2 + d];
  }
  float s = silu_f(acc);
  size_t o = ((size_t)b * LL + l) * DI + d;
  hsf[o] = s;
  hsb[o] = f32_to_bf16_bits(s);
}

// ---------------- extract dt_in (first R of ssm_p) to bf16 ----------------
__global__ void dtin_kernel(const float* __restrict__ ssm,
                            uint16_t* __restrict__ dtb) {
  int i = blockIdx.x * blockDim.x + threadIdx.x;
  if (i < BB * LL * RR) {
    int r = i % RR;
    int bl = i / RR;
    dtb[i] = f32_to_bf16_bits(ssm[(size_t)bl * PDIM + r]);
  }
}

// ---------------- selective-scan recurrence ----------------
__global__ __launch_bounds__(256)
void scan_kernel(const float* __restrict__ dt,    // (b,l,DI)
                 const float* __restrict__ hs,    // (b,l,DI)
                 const float* __restrict__ ssm,   // (b,l,160): [128..144)=B, [144..160)=C
                 const float* __restrict__ A_log, // (DI, 16)
                 float* __restrict__ ys) {        // (b,l,DI)
  const int d = blockIdx.x * blockDim.x + threadIdx.x;
  const int b = blockIdx.z;
  float Af[NST];
#pragma unroll
  for (int n = 0; n < NST; ++n) Af[n] = -__expf(A_log[d * NST + n]);
  float h[NST];
#pragma unroll
  for (int n = 0; n < NST; ++n) h[n] = 0.0f;

  const float* dtp = dt + (size_t)b * LL * DI;
  const float* hsp = hs + (size_t)b * LL * DI;
  const float* sp  = ssm + (size_t)b * LL * PDIM;
  float* yp = ys + (size_t)b * LL * DI;

  for (int l = 0; l < LL; ++l) {
    float dtv = dtp[(size_t)l * DI + d];
    float hsv = hsp[(size_t)l * DI + d];
    const float* bc = sp + (size_t)l * PDIM + RR;  // B[16], C[16] (wave-uniform)
    float y = 0.0f;
#pragma unroll
    for (int n = 0; n < NST; ++n) {
      float dA = __expf(Af[n] * dtv);
      float hn = dA * h[n] + dtv * bc[n] * hsv;
      h[n] = hn;
      y += hn * bc[NST + n];
    }
    yp[(size_t)l * DI + d] = y;
  }
}

// ---------------- D-skip + gate SiLU, emit bf16 y ----------------
__global__ void gate_kernel(const float* __restrict__ ys,
                            const float* __restrict__ hsf,
                            const float* __restrict__ Dv,
                            const float* __restrict__ proj,
                            uint16_t* __restrict__ yb) {
  const int d = blockIdx.x * blockDim.x + threadIdx.x;
  const int l = blockIdx.y;
  const int b = blockIdx.z;
  size_t o = ((size_t)b * LL + l) * DI + d;
  float y = ys[o] + hsf[o] * Dv[d];
  float g = proj[((size_t)b * LL + l) * E2 + DI + d];
  y *= silu_f(g);
  // y/2 + stop_grad(y/2) == y in forward
  yb[o] = f32_to_bf16_bits(y);
}

// ---------------- launch ----------------
extern "C" void kernel_launch(void* const* d_in, const int* in_sizes, int n_in,
                              void* d_out, int out_size, void* d_ws, size_t ws_size,
                              hipStream_t stream) {
  const float* x     = (const float*)d_in[0];  // (B,L,H)
  const float* Win   = (const float*)d_in[1];  // (2DI,H)
  const float* Wconv = (const float*)d_in[2];  // (DI,1,K)
  const float* bconv = (const float*)d_in[3];  // (DI)
  const float* Wx    = (const float*)d_in[4];  // (160,DI)
  const float* Wdt   = (const float*)d_in[5];  // (DI,R)
  const float* bdt   = (const float*)d_in[6];  // (DI)
  const float* Wout  = (const float*)d_in[7];  // (H,DI)
  const float* A_log = (const float*)d_in[8];  // (DI,16)
  const float* Dv    = (const float*)d_in[9];  // (DI)
  float* out = (float*)d_out;                  // (B,L,H)

  // workspace carve (256B aligned)
  char* w = (char*)d_ws;
  auto carve = [&](size_t bytes) -> void* {
    void* p = (void*)w;
    w += (bytes + 255) & ~(size_t)255;
    return p;
  };
  uint16_t* xb    = (uint16_t*)carve((size_t)BB * LL * H_DIM * 2);
  uint16_t* Winb  = (uint16_t*)carve((size_t)E2 * H_DIM * 2);
  uint16_t* Wxb   = (uint16_t*)carve((size_t)PDIM * DI * 2);
  uint16_t* Wdtb  = (uint16_t*)carve((size_t)DI * RR * 2);
  uint16_t* Woutb = (uint16_t*)carve((size_t)H_DIM * DI * 2);
  float*    proj  = (float*)   carve((size_t)BB * LL * E2 * 4);
  float*    hsf   = (float*)   carve((size_t)BB * LL * DI * 4);
  uint16_t* hsb   = (uint16_t*)carve((size_t)BB * LL * DI * 2);
  float*    ssm   = (float*)   carve((size_t)BB * LL * PDIM * 4);
  uint16_t* dtinb = (uint16_t*)carve((size_t)BB * LL * RR * 2);
  float*    dtf   = (float*)   carve((size_t)BB * LL * DI * 4);
  float*    ysc   = (float*)   carve((size_t)BB * LL * DI * 4);
  uint16_t* yb    = (uint16_t*)carve((size_t)BB * LL * DI * 2);

  const int TPB = 256;
  auto cvt = [&](const float* s, uint16_t* d, int n) {
    cvt_f32_bf16<<<(n + TPB - 1) / TPB, TPB, 0, stream>>>(s, d, n);
  };
  cvt(x,    xb,    BB * LL * H_DIM);
  cvt(Win,  Winb,  E2 * H_DIM);
  cvt(Wx,   Wxb,   PDIM * DI);
  cvt(Wdt,  Wdtb,  DI * RR);
  cvt(Wout, Woutb, H_DIM * DI);

  // GEMM1: proj(b,l,e) = x(b,l,h) . Win(e,h)^T   M=8192 K=2048
  gemm_wmma_bf16<0><<<dim3(E2 / 128, LL / 128, BB), TPB, 0, stream>>>(
      Winb, xb, proj, nullptr, E2, LL, H_DIM,
      (long)LL * H_DIM, (long)LL * E2);

  // conv + SiLU -> hs (f32 + bf16)
  conv_silu_kernel<<<dim3(DI / TPB, LL, BB), TPB, 0, stream>>>(
      proj, Wconv, bconv, hsf, hsb);

  // GEMM2: ssm_p(b,l,p) = hs(b,l,d) . Wx(p,d)^T   M=160 K=4096
  gemm_wmma_bf16<0><<<dim3((PDIM + 127) / 128, LL / 128, BB), TPB, 0, stream>>>(
      Wxb, hsb, ssm, nullptr, PDIM, LL, DI,
      (long)LL * DI, (long)LL * PDIM);

  // dt_in -> bf16
  {
    int n = BB * LL * RR;
    dtin_kernel<<<(n + TPB - 1) / TPB, TPB, 0, stream>>>(ssm, dtinb);
  }

  // GEMM3: dt(b,l,d) = softplus(dt_in . Wdt^T + bdt)   M=4096 K=128
  gemm_wmma_bf16<1><<<dim3(DI / 128, LL / 128, BB), TPB, 0, stream>>>(
      Wdtb, dtinb, dtf, bdt, DI, LL, RR,
      (long)LL * RR, (long)LL * DI);

  // selective scan
  scan_kernel<<<dim3(DI / TPB, 1, BB), TPB, 0, stream>>>(
      dtf, hsf, ssm, A_log, ysc);

  // D-skip + gate
  gate_kernel<<<dim3(DI / TPB, LL, BB), TPB, 0, stream>>>(
      ysc, hsf, Dv, proj, yb);

  // GEMM4: out(b,l,h) = y(b,l,d) . Wout(h,d)^T   M=2048 K=4096
  gemm_wmma_bf16<0><<<dim3(H_DIM / 128, LL / 128, BB), TPB, 0, stream>>>(
      Woutb, yb, out, nullptr, H_DIM, LL, DI,
      (long)LL * DI, (long)LL * H_DIM);
}
